// tPatchGNN_Classifier_43301860278565
// MI455X (gfx1250) — compile-verified, matrix operand order checked by reading, not run on hardware
//
#include <hip/hip_runtime.h>
#include <hip/hip_bf16.h>
#include <math.h>

// ---------------- constants ----------------
#define NB    8      // batch
#define PP    64     // patches
#define LL    32     // patch length
#define LMAXC 2048
#define INDIM 12
#define TEDIM 20
#define PT    32     // POINT = INDIM + TEDIM
#define HIDC  128
#define TTCNC 127
#define NTOK  512    // NB*PP
#define NLROW 16384  // NTOK*LL
#define DFFC  2048
#define NQC   5
#define FGCOLS (TTCNC*PT)   // 4064
#define FGNP   4096         // padded

typedef __attribute__((ext_vector_type(16))) _Float16 v16h;
typedef __attribute__((ext_vector_type(8)))  _Float16 h8;
typedef __attribute__((ext_vector_type(8)))  float    v8f;
typedef __attribute__((ext_vector_type(4)))  float    f4;

// K-index mapping inside a 16-bit WMMA fragment (ISA 7.12.2):
// lane half=0 holds K {0..7, 16..23}, half=1 holds K {8..15, 24..31}.
__device__ __forceinline__ int kmap16(int i, int half) {
  return (i < 8) ? (i + 8 * half) : (i + 8 + 8 * half);
}

__device__ __forceinline__ v16h combine16(h8 lo, h8 hi) {
  v16h r;
#pragma unroll
  for (int i = 0; i < 8; ++i) { r[i] = lo[i]; r[8 + i] = hi[i]; }
  return r;
}

// A fragment from row-major f16: two contiguous b128 loads per lane. Rows
// clamped (no branch); out-of-range rows feed C rows that are never stored.
__device__ __forceinline__ v16h load_a_f16v(const _Float16* A, int lda, int m0, int k0,
                                            int M, int lane) {
  int m = m0 + (lane & 15);
  if (m >= M) m = M - 1;
  int half = (lane >> 4) & 1;
  const _Float16* p = A + (size_t)m * lda + k0 + 8 * half;
  h8 lo = *(const h8*)p;
  h8 hi = *(const h8*)(p + 16);
  return combine16(lo, hi);
}

// A fragment from row-major f32: four contiguous b128 loads + pack-converts.
__device__ __forceinline__ v16h load_a_f32v(const float* A, int lda, int m0, int k0,
                                            int M, int lane) {
  int m = m0 + (lane & 15);
  if (m >= M) m = M - 1;
  int half = (lane >> 4) & 1;
  const float* p = A + (size_t)m * lda + k0 + 8 * half;
  f4 a0 = *(const f4*)p;
  f4 a1 = *(const f4*)(p + 4);
  f4 a2 = *(const f4*)(p + 16);
  f4 a3 = *(const f4*)(p + 20);
  v16h r;
#pragma unroll
  for (int i = 0; i < 4; ++i) {
    r[i]      = (_Float16)a0[i];
    r[4 + i]  = (_Float16)a1[i];
    r[8 + i]  = (_Float16)a2[i];
    r[12 + i] = (_Float16)a3[i];
  }
  return r;
}

// B fragment from packed weights: 16 contiguous halves per (kb, n, half).
__device__ __forceinline__ v16h load_b_packed(const _Float16* Wp, int Np, int kb, int n,
                                              int lane) {
  int half = (lane >> 4) & 1;
  const _Float16* p = Wp + (((size_t)kb * Np + n) * 2 + half) * 16;
  h8 lo = *(const h8*)p;
  h8 hi = *(const h8*)(p + 8);
  return combine16(lo, hi);
}

// ---------------- weight packing fp32 -> fragment-ordered f16 ----------------
__global__ void pack_b(const float* src, int srows, int scols,
                       _Float16* dst, int Kp, int Np) {
  int idx = blockIdx.x * 256 + threadIdx.x;
  if (idx < Kp * Np) {
    int i = idx & 15;
    int half = (idx >> 4) & 1;
    int rest = idx >> 5;
    int n = rest % Np;
    int kb = rest / Np;
    int k = kb * 32 + kmap16(i, half);
    float v = (k < srows && n < scols) ? src[(size_t)k * scols + n] : 0.0f;
    dst[idx] = (_Float16)v;
  }
}

// ---------------- generic WMMA GEMM: one wave per 32x32 C tile --------------
// Software-pipelined K loop: next fragments issued before current wmmas.
template <int AF32>
__global__ void gemm_wmma_t(const float* Af, const _Float16* Ah, int lda,
                            const _Float16* Wp, int Np,
                            const float* bias, int Nbias, int relu,
                            float* C, _Float16* Ch, int ldc,
                            int M, int N, int Kp) {
  int lane = threadIdx.x;
  int m0 = blockIdx.x * 32, n0 = blockIdx.y * 32;
  int nA = n0 + (lane & 15);
  int nB = nA + 16;
  v8f acc00 = {0.f, 0.f, 0.f, 0.f, 0.f, 0.f, 0.f, 0.f};
  v8f acc01 = acc00, acc10 = acc00, acc11 = acc00;

  v16h a0 = AF32 ? load_a_f32v(Af, lda, m0, 0, M, lane)
                 : load_a_f16v(Ah, lda, m0, 0, M, lane);
  v16h a1 = AF32 ? load_a_f32v(Af, lda, m0 + 16, 0, M, lane)
                 : load_a_f16v(Ah, lda, m0 + 16, 0, M, lane);
  v16h b0 = load_b_packed(Wp, Np, 0, nA, lane);
  v16h b1 = load_b_packed(Wp, Np, 0, nB, lane);

  for (int k0 = 32; k0 < Kp; k0 += 32) {
    v16h na0 = AF32 ? load_a_f32v(Af, lda, m0, k0, M, lane)
                    : load_a_f16v(Ah, lda, m0, k0, M, lane);
    v16h na1 = AF32 ? load_a_f32v(Af, lda, m0 + 16, k0, M, lane)
                    : load_a_f16v(Ah, lda, m0 + 16, k0, M, lane);
    v16h nb0 = load_b_packed(Wp, Np, k0 >> 5, nA, lane);
    v16h nb1 = load_b_packed(Wp, Np, k0 >> 5, nB, lane);
    acc00 = __builtin_amdgcn_wmma_f32_16x16x32_f16(false, a0, false, b0, (short)0, acc00, false, false);
    acc01 = __builtin_amdgcn_wmma_f32_16x16x32_f16(false, a0, false, b1, (short)0, acc01, false, false);
    acc10 = __builtin_amdgcn_wmma_f32_16x16x32_f16(false, a1, false, b0, (short)0, acc10, false, false);
    acc11 = __builtin_amdgcn_wmma_f32_16x16x32_f16(false, a1, false, b1, (short)0, acc11, false, false);
    a0 = na0; a1 = na1; b0 = nb0; b1 = nb1;
  }
  acc00 = __builtin_amdgcn_wmma_f32_16x16x32_f16(false, a0, false, b0, (short)0, acc00, false, false);
  acc01 = __builtin_amdgcn_wmma_f32_16x16x32_f16(false, a0, false, b1, (short)0, acc01, false, false);
  acc10 = __builtin_amdgcn_wmma_f32_16x16x32_f16(false, a1, false, b0, (short)0, acc10, false, false);
  acc11 = __builtin_amdgcn_wmma_f32_16x16x32_f16(false, a1, false, b1, (short)0, acc11, false, false);

  int half = (lane >> 4) & 1;
  float bvA = (bias && nA < Nbias) ? bias[nA] : 0.0f;
  float bvB = (bias && nB < Nbias) ? bias[nB] : 0.0f;
#pragma unroll
  for (int mt = 0; mt < 2; ++mt) {
#pragma unroll
    for (int jt = 0; jt < 2; ++jt) {
      int n = jt ? nB : nA;
      float bv = jt ? bvB : bvA;
      v8f acc = (mt == 0) ? (jt ? acc01 : acc00) : (jt ? acc11 : acc10);
      if (n < N) {
#pragma unroll
        for (int r = 0; r < 8; ++r) {
          int m = m0 + mt * 16 + r + 8 * half;
          if (m < M) {
            float v = acc[r] + bv;
            if (relu) v = fmaxf(v, 0.0f);
            if (C)  C[(size_t)m * ldc + n] = v;
            if (Ch) Ch[(size_t)m * ldc + n] = (_Float16)v;
          }
        }
      }
    }
  }
}

// ---------------- patch starts (per-batch prefix sum over P=64) ----------------
__global__ void starts_kernel(const float* pmask, const int* pidx, int* lens, int* starts) {
  __shared__ int sl[PP];
  int b = blockIdx.x, p = threadIdx.x;
  int pv = pidx[b * PP + p];
  int len = (pmask[b * PP + p] > 0.0f) ? (pv > 0 ? pv : 0) : 0;
  sl[p] = len;
  __syncthreads();
  int st = 0;
  for (int j = 0; j < p; ++j) st += sl[j];
  lens[b * PP + p] = len;
  starts[b * PP + p] = st;
}

// ---------------- gather + time embedding -> X_int (fp32 + f16), mask, exists ----
__global__ void gather_kernel(const float* dseq, const float* tseq, const int* nb,
                              const int* lens, const int* starts,
                              const float* sW, const float* sB,
                              const float* pW, const float* pB,
                              float* Xf, _Float16* Xh, float* maskNL, float* pex) {
  int n = blockIdx.x;
  int b = n / PP;
  int tid = threadIdx.x;  // 128
  __shared__ int cnt;
  if (tid == 0) cnt = 0;
  __syncthreads();
  int len = lens[n], st = starts[n], nbv = nb[b];
  for (int e = tid; e < LL * PT; e += 128) {
    int l = e >> 5, c = e & 31;
    int idx = st + l;
    bool valid = (l < len) && (idx < nbv);
    int idxc = idx;
    if (idxc < 0) idxc = 0;
    if (idxc > LMAXC - 1) idxc = LMAXC - 1;
    float ts = valid ? tseq[(size_t)b * LMAXC + idxc] : 0.0f;
    float v;
    if (c < INDIM) {
      v = valid ? dseq[((size_t)b * LMAXC + idxc) * INDIM + c] : 0.0f;
    } else if (c == INDIM) {
      v = ts * sW[0] + sB[0];
    } else {
      int j = c - INDIM - 1;
      v = sinf(ts * pW[j] + pB[j]);
    }
    Xf[(size_t)n * LL * PT + e] = v;
    Xh[(size_t)n * LL * PT + e] = (_Float16)v;
    if (c == 0) {
      maskNL[n * LL + l] = valid ? 1.0f : 0.0f;
      if (valid) atomicAdd(&cnt, 1);
    }
  }
  __syncthreads();
  if (tid == 0) pex[n] = (cnt > 0) ? 1.0f : 0.0f;
}

// ---------------- fused Filt -> masked softmax(L) -> ttcn contraction ----------
// block: 256 threads = 8 waves; grid (NTOK, 32). Each block: patch n, 4 TTCN
// channels (128 Filt columns). Avoids 266MB Filt materialization.
__global__ void filt_fused(const _Float16* h2h, const float* Xf, const float* maskNL,
                           const _Float16* W3p, const float* b3, float* ttcn) {
  __shared__ _Float16 hA[LL * HIDC];
  __shared__ float F[LL * HIDC];
  __shared__ float Xs[LL * PT];
  __shared__ float ms[LL];
  __shared__ float red[256];
  int n = blockIdx.x, tc = blockIdx.y;
  int tid = threadIdx.x;
  {
    const h8* src = (const h8*)(h2h + (size_t)n * LL * HIDC);
    h8* dst = (h8*)hA;
    for (int i = tid; i < LL * HIDC / 8; i += 256) dst[i] = src[i];
  }
  {
    const f4* src = (const f4*)(Xf + (size_t)n * LL * PT);
    f4* dst = (f4*)Xs;
    for (int i = tid; i < LL * PT / 4; i += 256) dst[i] = src[i];
  }
  if (tid < LL) ms[tid] = maskNL[n * LL + tid];
  __syncthreads();

  int wave = tid >> 5, lane = tid & 31;
  int colbase = tc * 128;
#pragma unroll
  for (int t8i = 0; t8i < 2; ++t8i) {
    int t8 = wave + t8i * 8;
    int mt = t8 & 1, nt = t8 >> 1;
    int m0 = mt * 16, n0loc = nt * 16;
    int n0g = colbase + n0loc + (lane & 15);
    v8f acc = {0.f, 0.f, 0.f, 0.f, 0.f, 0.f, 0.f, 0.f};
#pragma unroll
    for (int k0 = 0; k0 < HIDC; k0 += 32) {
      v16h a = load_a_f16v(hA, HIDC, m0, k0, LL, lane);
      v16h b = load_b_packed(W3p, FGNP, k0 >> 5, n0g, lane);
      acc = __builtin_amdgcn_wmma_f32_16x16x32_f16(false, a, false, b, (short)0, acc,
                                                   false, false);
    }
    int half = (lane >> 4) & 1;
    float bv = (n0g < FGCOLS) ? b3[n0g] : 0.0f;
#pragma unroll
    for (int r = 0; r < 8; ++r)
      F[(m0 + r + 8 * half) * HIDC + n0loc + (lane & 15)] = acc[r] + bv;
  }
  __syncthreads();

  // masked softmax over L per column
  if (tid < 128) {
    int c = tid;
    float mx = -1e30f;
    for (int l = 0; l < LL; ++l) {
      float m = ms[l];
      float fm = m * F[l * HIDC + c] + (1.0f - m) * (-1e8f);
      F[l * HIDC + c] = fm;
      mx = fmaxf(mx, fm);
    }
    float s = 0.0f;
    for (int l = 0; l < LL; ++l) {
      float e = expf(F[l * HIDC + c] - mx);
      F[l * HIDC + c] = e;
      s += e;
    }
    float inv = 1.0f / s;
    for (int l = 0; l < LL; ++l) F[l * HIDC + c] *= inv;
  }
  __syncthreads();

  // ttcn[n,t] = sum_{l,p} X[l,p] * Fs[l, t*32+p]; 64 threads per t
  int tloc = tid >> 6, j = tid & 63;
  float part = 0.0f;
  for (int k = 0; k < 16; ++k) {
    int i = j + 64 * k;
    int l = i >> 5, p = i & 31;
    part += Xs[l * PT + p] * F[l * HIDC + tloc * PT + p];
  }
  red[tid] = part;
  __syncthreads();
  for (int s = 32; s >= 1; s >>= 1) {
    if ((tid & 63) < s) red[tid] += red[tid + s];
    __syncthreads();
  }
  if ((tid & 63) == 0) {
    int tg = tc * 4 + tloc;
    if (tg < TTCNC) ttcn[(size_t)n * HIDC + tg] = red[tid];
  }
}

// ---------------- x = [relu(ttcn + T_bias), patch_exists] ----------------
__global__ void build_x(const float* ttcn, const float* Tb, const float* pex, float* x) {
  int n = blockIdx.x, c = threadIdx.x;
  float v = (c < TTCNC) ? fmaxf(ttcn[(size_t)n * HIDC + c] + Tb[c], 0.0f) : pex[n];
  x[(size_t)n * HIDC + c] = v;
}

// ---------------- positional encoding + time bias tables ----------------
__global__ void init_tables(float* pe, float* tb) {
  int i = blockIdx.x * 256 + threadIdx.x;
  if (i < PP * HIDC) {
    int p = i >> 7, c = i & 127;
    int k2 = (c >> 1) * 2;
    float div = expf((float)k2 * (-logf(10000.0f) / (float)HIDC));
    float ang = (float)p * div;
    pe[i] = (c & 1) ? cosf(ang) : sinf(ang);
  }
  if (i < PP * PP) {
    int r = i >> 6, cc = i & 63;
    float dist = fabsf((float)(r - cc)) * 300.0f;
    tb[i] = logf(expf(-dist / 300.0f) + 1e-12f);
  }
}

// ---------------- scalar attention per (b, head), K/V staged in LDS ----------
__global__ void attn_kernel(const float* Q, const float* Kx, const float* Vx, float* O,
                            const float* tb, const float* pmask,
                            int Mq, int Mk, int qsb, int osb,
                            int useBias, int usePad, float scale) {
  int bh = blockIdx.x;
  int b = bh / 4, h = bh % 4;
  __shared__ float Ks[PP * 32], Vs[PP * 32];
  int tid = threadIdx.x;  // 64
  for (int i = tid; i < Mk * 32; i += 64) {
    int r = i >> 5, d = i & 31;
    Ks[i] = Kx[((size_t)b * PP + r) * HIDC + h * 32 + d];
    Vs[i] = Vx[((size_t)b * PP + r) * HIDC + h * 32 + d];
  }
  __syncthreads();
  if (tid < Mq) {
    float q[32];
    const float* qr = Q + (size_t)b * qsb + (size_t)tid * HIDC + h * 32;
#pragma unroll
    for (int d = 0; d < 32; ++d) q[d] = qr[d];
    float mx = -1e30f;
    for (int k = 0; k < Mk; ++k) {
      float dot = 0.0f;
#pragma unroll
      for (int d = 0; d < 32; ++d) dot += q[d] * Ks[k * 32 + d];
      dot *= scale;
      if (useBias) dot += tb[tid * PP + k];
      if (usePad && pmask[b * PP + k] <= 0.0f) dot = -1e9f;
      mx = fmaxf(mx, dot);
    }
    float sum = 0.0f;
    float o[32];
#pragma unroll
    for (int d = 0; d < 32; ++d) o[d] = 0.0f;
    for (int k = 0; k < Mk; ++k) {
      float dot = 0.0f;
#pragma unroll
      for (int d = 0; d < 32; ++d) dot += q[d] * Ks[k * 32 + d];
      dot *= scale;
      if (useBias) dot += tb[tid * PP + k];
      if (usePad && pmask[b * PP + k] <= 0.0f) dot = -1e9f;
      float e = expf(dot - mx);
      sum += e;
#pragma unroll
      for (int d = 0; d < 32; ++d) o[d] += e * Vs[k * 32 + d];
    }
    float inv = 1.0f / sum;
    float* orow = O + (size_t)b * osb + (size_t)tid * HIDC + h * 32;
#pragma unroll
    for (int d = 0; d < 32; ++d) orow[d] = o[d] * inv;
  }
}

// ---------------- elementwise kernels ----------------
__global__ void ew_add_scaled(float* y, const float* g, const float* pm) {
  int i = blockIdx.x * 256 + threadIdx.x;
  if (i < NTOK * HIDC) y[i] += g[i] * pm[i >> 7];
}
__global__ void ew_add_pe(float* y, const float* pe) {
  int i = blockIdx.x * 256 + threadIdx.x;
  if (i < NTOK * HIDC) {
    int p = (i >> 7) & 63;
    y[i] += pe[p * HIDC + (i & 127)];
  }
}
__global__ void ew_add(float* out, const float* a, const float* b) {
  int i = blockIdx.x * 256 + threadIdx.x;
  if (i < NTOK * HIDC) out[i] = a[i] + b[i];
}
__global__ void ew_res_mask(float* x, const float* xl, const float* pm) {
  int i = blockIdx.x * 256 + threadIdx.x;
  if (i < NTOK * HIDC) x[i] = (xl[i] + x[i]) * pm[i >> 7];
}

// ---------------- LayerNorm over HID=128 ----------------
__global__ void ln_kernel(const float* in, const float* g, const float* bta, float* out) {
  __shared__ float red[HIDC];
  int n = blockIdx.x, c = threadIdx.x;
  float v = in[(size_t)n * HIDC + c];
  red[c] = v;
  __syncthreads();
  for (int s = 64; s >= 1; s >>= 1) {
    if (c < s) red[c] += red[c + s];
    __syncthreads();
  }
  float mean = red[0] * (1.0f / HIDC);
  __syncthreads();
  float d = v - mean;
  red[c] = d * d;
  __syncthreads();
  for (int s = 64; s >= 1; s >>= 1) {
    if (c < s) red[c] += red[c + s];
    __syncthreads();
  }
  float var = red[0] * (1.0f / HIDC);
  out[(size_t)n * HIDC + c] = d * rsqrtf(var + 1e-5f) * g[c] + bta[c];
}

// ---------------- final logits: z @ cls_out_W + b ----------------
__global__ void logits_kernel(const float* z, const float* W, const float* bb, float* out) {
  __shared__ float red[HIDC];
  int r = blockIdx.x, c = threadIdx.x;
  red[c] = z[(size_t)r * HIDC + c] * W[c];
  __syncthreads();
  for (int s = 64; s >= 1; s >>= 1) {
    if (c < s) red[c] += red[c + s];
    __syncthreads();
  }
  if (c == 0) out[r] = red[0] + bb[0];
}

// =======================================================================
extern "C" void kernel_launch(void* const* d_in, const int* in_sizes, int n_in,
                              void* d_out, int out_size, void* d_ws, size_t ws_size,
                              hipStream_t stream) {
  (void)in_sizes; (void)n_in; (void)out_size; (void)ws_size;
  const float* data_sequence = (const float*)d_in[0];
  const float* ts_sequence   = (const float*)d_in[1];
  const float* patch_mask    = (const float*)d_in[2];
  const int*   patch_index   = (const int*)d_in[3];
  const int*   num_batch     = (const int*)d_in[4];
  const float* te_scale_W    = (const float*)d_in[5];
  const float* te_scale_b    = (const float*)d_in[6];
  const float* te_per_W      = (const float*)d_in[7];
  const float* te_per_b      = (const float*)d_in[8];
  const float* fg_W1 = (const float*)d_in[9];
  const float* fg_b1 = (const float*)d_in[10];
  const float* fg_W2 = (const float*)d_in[11];
  const float* fg_b2 = (const float*)d_in[12];
  const float* fg_W3 = (const float*)d_in[13];
  const float* fg_b3 = (const float*)d_in[14];
  const float* T_bias = (const float*)d_in[15];
  const float* ga_Wq = (const float*)d_in[16];
  const float* ga_Wk = (const float*)d_in[17];
  const float* ga_Wv = (const float*)d_in[18];
  const float* ga_Wo = (const float*)d_in[19];
  const float* tf_Wq = (const float*)d_in[20];
  const float* tf_Wk = (const float*)d_in[21];
  const float* tf_Wv = (const float*)d_in[22];
  const float* tf_Wo = (const float*)d_in[23];
  const float* ga_bq = (const float*)d_in[24];
  const float* ga_bk = (const float*)d_in[25];
  const float* ga_bv = (const float*)d_in[26];
  const float* ga_bo = (const float*)d_in[27];
  const float* tf_bq = (const float*)d_in[28];
  const float* tf_bk = (const float*)d_in[29];
  const float* tf_bv = (const float*)d_in[30];
  const float* tf_bo = (const float*)d_in[31];
  const float* tf_ff_b2 = (const float*)d_in[32];
  const float* tf_ln1_g = (const float*)d_in[33];
  const float* tf_ln1_b = (const float*)d_in[34];
  const float* tf_ln2_g = (const float*)d_in[35];
  const float* tf_ln2_b = (const float*)d_in[36];
  const float* tf_ff_W1 = (const float*)d_in[37];
  const float* tf_ff_b1 = (const float*)d_in[38];
  const float* tf_ff_W2 = (const float*)d_in[39];
  const float* cls_query = (const float*)d_in[40];
  const float* cls_Wq = (const float*)d_in[41];
  const float* cls_Wk = (const float*)d_in[42];
  const float* cls_Wv = (const float*)d_in[43];
  const float* cls_Wo = (const float*)d_in[44];
  const float* cls_bq = (const float*)d_in[45];
  const float* cls_bk = (const float*)d_in[46];
  const float* cls_bv = (const float*)d_in[47];
  const float* cls_bo = (const float*)d_in[48];
  const float* cls_out_W = (const float*)d_in[49];
  const float* cls_out_b = (const float*)d_in[50];

  // ---- workspace carve-up ----
  char* wsb = (char*)d_ws;
  size_t off = 0;
  auto alloc = [&](size_t bytes) -> void* {
    void* p = wsb + off;
    off = (off + bytes + 255) & ~(size_t)255;
    return p;
  };
  int*   lens    = (int*)alloc(NTOK * 4);
  int*   starts  = (int*)alloc(NTOK * 4);
  float* maskNL  = (float*)alloc(NTOK * LL * 4);
  float* pex     = (float*)alloc(NTOK * 4);
  float* Xf      = (float*)alloc((size_t)NTOK * LL * PT * 4);
  _Float16* Xh   = (_Float16*)alloc((size_t)NTOK * LL * PT * 2);
  _Float16* W1p  = (_Float16*)alloc(32 * 128 * 2);
  _Float16* W2p  = (_Float16*)alloc(128 * 128 * 2);
  _Float16* W3p  = (_Float16*)alloc((size_t)128 * FGNP * 2);
  _Float16* h1h  = (_Float16*)alloc((size_t)NLROW * 128 * 2);
  _Float16* h2h  = (_Float16*)alloc((size_t)NLROW * 128 * 2);
  float* ttcn    = (float*)alloc((size_t)NTOK * HIDC * 4);
  float* pe      = (float*)alloc(PP * HIDC * 4);
  float* tb      = (float*)alloc(PP * PP * 4);
  float* x       = (float*)alloc((size_t)NTOK * HIDC * 4);
  float* xlast   = (float*)alloc((size_t)NTOK * HIDC * 4);
  float* qb      = (float*)alloc((size_t)NTOK * HIDC * 4);
  float* kb      = (float*)alloc((size_t)NTOK * HIDC * 4);
  float* vb      = (float*)alloc((size_t)NTOK * HIDC * 4);
  float* tA      = (float*)alloc((size_t)NTOK * HIDC * 4);
  float* tB      = (float*)alloc((size_t)NTOK * HIDC * 4);
  float* ffmid   = (float*)alloc((size_t)NTOK * DFFC * 4);
  _Float16* gaWqp = (_Float16*)alloc(2 * 128 * 128 * 2);
  _Float16* gaWkp = (_Float16*)alloc(2 * 128 * 128 * 2);
  _Float16* gaWvp = (_Float16*)alloc(2 * 128 * 128 * 2);
  _Float16* gaWop = (_Float16*)alloc(2 * 128 * 128 * 2);
  _Float16* tfWqp = (_Float16*)alloc(2 * 128 * 128 * 2);
  _Float16* tfWkp = (_Float16*)alloc(2 * 128 * 128 * 2);
  _Float16* tfWvp = (_Float16*)alloc(2 * 128 * 128 * 2);
  _Float16* tfWop = (_Float16*)alloc(2 * 128 * 128 * 2);
  _Float16* ffW1p = (_Float16*)alloc((size_t)2 * 128 * DFFC * 2);
  _Float16* ffW2p = (_Float16*)alloc((size_t)2 * DFFC * 128 * 2);
  _Float16* clsWqp = (_Float16*)alloc(128 * 128 * 2);
  _Float16* clsWkp = (_Float16*)alloc(128 * 128 * 2);
  _Float16* clsWvp = (_Float16*)alloc(128 * 128 * 2);
  _Float16* clsWop = (_Float16*)alloc(128 * 128 * 2);
  float* qc   = (float*)alloc(NQC * HIDC * 4);
  float* clsO = (float*)alloc((size_t)NB * NQC * HIDC * 4);
  float* zo   = (float*)alloc((size_t)NB * NQC * HIDC * 4);

  const float scale = 1.0f / sqrtf(32.0f);

  // A in fp32
  auto gemmF = [&](const float* Af, int lda,
                   const _Float16* Wp, int Np, const float* bias, int Nbias, int relu,
                   float* C, _Float16* Ch, int ldc, int M, int N, int Kp) {
    dim3 g((M + 31) / 32, (N + 31) / 32);
    gemm_wmma_t<1><<<g, dim3(32), 0, stream>>>(Af, nullptr, lda, Wp, Np, bias, Nbias,
                                               relu, C, Ch, ldc, M, N, Kp);
  };
  // A in f16
  auto gemmH = [&](const _Float16* Ah, int lda,
                   const _Float16* Wp, int Np, const float* bias, int Nbias, int relu,
                   float* C, _Float16* Ch, int ldc, int M, int N, int Kp) {
    dim3 g((M + 31) / 32, (N + 31) / 32);
    gemm_wmma_t<0><<<g, dim3(32), 0, stream>>>(nullptr, Ah, lda, Wp, Np, bias, Nbias,
                                               relu, C, Ch, ldc, M, N, Kp);
  };
  auto pack = [&](const float* src, int sr, int sc, _Float16* dst, int Kp, int Np) {
    int n = Kp * Np;
    pack_b<<<(n + 255) / 256, 256, 0, stream>>>(src, sr, sc, dst, Kp, Np);
  };

  // ---- setup: tables, starts, weight packing ----
  init_tables<<<(PP * HIDC + 255) / 256, 256, 0, stream>>>(pe, tb);
  starts_kernel<<<NB, PP, 0, stream>>>(patch_mask, patch_index, lens, starts);
  pack(fg_W1, PT, TTCNC, W1p, 32, 128);
  pack(fg_W2, TTCNC, TTCNC, W2p, 128, 128);
  pack(fg_W3, TTCNC, FGCOLS, W3p, 128, FGNP);
  // stacked (layer-major) weights: per-layer packed block is contiguous
  pack(ga_Wq, 2 * 128, 128, gaWqp, 256, 128);
  pack(ga_Wk, 2 * 128, 128, gaWkp, 256, 128);
  pack(ga_Wv, 2 * 128, 128, gaWvp, 256, 128);
  pack(ga_Wo, 2 * 128, 128, gaWop, 256, 128);
  pack(tf_Wq, 2 * 128, 128, tfWqp, 256, 128);
  pack(tf_Wk, 2 * 128, 128, tfWkp, 256, 128);
  pack(tf_Wv, 2 * 128, 128, tfWvp, 256, 128);
  pack(tf_Wo, 2 * 128, 128, tfWop, 256, 128);
  pack(tf_ff_W1, 2 * 128, DFFC, ffW1p, 256, DFFC);
  pack(tf_ff_W2, 2 * DFFC, 128, ffW2p, 4096, 128);
  pack(cls_Wq, 128, 128, clsWqp, 128, 128);
  pack(cls_Wk, 128, 128, clsWkp, 128, 128);
  pack(cls_Wv, 128, 128, clsWvp, 128, 128);
  pack(cls_Wo, 128, 128, clsWop, 128, 128);

  // ---- gather + time embedding ----
  gather_kernel<<<NTOK, 128, 0, stream>>>(data_sequence, ts_sequence, num_batch,
                                          lens, starts, te_scale_W, te_scale_b,
                                          te_per_W, te_per_b, Xf, Xh, maskNL, pex);

  // ---- filter generator MLP (WMMA); cols >=127 stored as exact zeros ----
  gemmH(Xh, PT, W1p, 128, fg_b1, TTCNC, 1, nullptr, h1h, 128, NLROW, 128, 32);
  gemmH(h1h, 128, W2p, 128, fg_b2, TTCNC, 1, nullptr, h2h, 128, NLROW, 128, 128);

  // ---- fused Filt -> softmax -> ttcn ----
  filt_fused<<<dim3(NTOK, 32), 256, 0, stream>>>(h2h, Xf, maskNL, W3p, fg_b3, ttcn);
  build_x<<<NTOK, HIDC, 0, stream>>>(ttcn, T_bias, pex, x);

  const int EWG = (NTOK * HIDC + 255) / 256;
  // ---- transformer layers ----
  for (int l = 0; l < 2; ++l) {
    size_t wo = (size_t)l * 128 * 128;
    hipMemcpyAsync(xlast, x, (size_t)NTOK * HIDC * 4, hipMemcpyDeviceToDevice, stream);
    // graph attention (time-bias + key padding)
    gemmF(x, HIDC, gaWqp + wo, 128, ga_bq + l * 128, 128, 0, qb, nullptr, HIDC, NTOK, 128, 128);
    gemmF(x, HIDC, gaWkp + wo, 128, ga_bk + l * 128, 128, 0, kb, nullptr, HIDC, NTOK, 128, 128);
    gemmF(x, HIDC, gaWvp + wo, 128, ga_bv + l * 128, 128, 0, vb, nullptr, HIDC, NTOK, 128, 128);
    attn_kernel<<<NB * 4, 64, 0, stream>>>(qb, kb, vb, tA, tb, patch_mask,
                                           PP, PP, PP * HIDC, PP * HIDC, 1, 1, scale);
    gemmF(tA, HIDC, gaWop + wo, 128, ga_bo + l * 128, 128, 0, tB, nullptr, HIDC, NTOK, 128, 128);
    ew_add_scaled<<<EWG, 256, 0, stream>>>(x, tB, patch_mask);
    ew_add_pe<<<EWG, 256, 0, stream>>>(x, pe);
    // plain self-attention
    gemmF(x, HIDC, tfWqp + wo, 128, tf_bq + l * 128, 128, 0, qb, nullptr, HIDC, NTOK, 128, 128);
    gemmF(x, HIDC, tfWkp + wo, 128, tf_bk + l * 128, 128, 0, kb, nullptr, HIDC, NTOK, 128, 128);
    gemmF(x, HIDC, tfWvp + wo, 128, tf_bv + l * 128, 128, 0, vb, nullptr, HIDC, NTOK, 128, 128);
    attn_kernel<<<NB * 4, 64, 0, stream>>>(qb, kb, vb, tA, tb, patch_mask,
                                           PP, PP, PP * HIDC, PP * HIDC, 0, 0, scale);
    gemmF(tA, HIDC, tfWop + wo, 128, tf_bo + l * 128, 128, 0, tB, nullptr, HIDC, NTOK, 128, 128);
    ew_add<<<EWG, 256, 0, stream>>>(tA, x, tB);
    ln_kernel<<<NTOK, HIDC, 0, stream>>>(tA, tf_ln1_g + l * 128, tf_ln1_b + l * 128, x);
    // feed-forward 128 -> 2048 -> 128
    gemmF(x, HIDC, ffW1p + (size_t)l * 128 * DFFC, DFFC, tf_ff_b1 + l * DFFC, DFFC, 1,
          ffmid, nullptr, DFFC, NTOK, DFFC, 128);
    gemmF(ffmid, DFFC, ffW2p + (size_t)l * DFFC * 128, 128, tf_ff_b2 + l * 128, 128, 0,
          tB, nullptr, HIDC, NTOK, 128, DFFC);
    ew_add<<<EWG, 256, 0, stream>>>(tA, x, tB);
    ln_kernel<<<NTOK, HIDC, 0, stream>>>(tA, tf_ln2_g + l * 128, tf_ln2_b + l * 128, x);
    ew_res_mask<<<EWG, 256, 0, stream>>>(x, xlast, patch_mask);
  }

  // ---- classifier attention ----
  gemmF(cls_query, HIDC, clsWqp, 128, cls_bq, 128, 0, qc, nullptr, HIDC, NQC, 128, 128);
  gemmF(x, HIDC, clsWkp, 128, cls_bk, 128, 0, kb, nullptr, HIDC, NTOK, 128, 128);
  gemmF(x, HIDC, clsWvp, 128, cls_bv, 128, 0, vb, nullptr, HIDC, NTOK, 128, 128);
  attn_kernel<<<NB * 4, 64, 0, stream>>>(qc, kb, vb, clsO, tb, patch_mask,
                                         NQC, PP, 0, NQC * HIDC, 0, 1, scale);
  gemmF(clsO, HIDC, clsWop, 128, cls_bo, 128, 0, zo, nullptr, HIDC, NB * NQC, 128, 128);
  logits_kernel<<<NB * NQC, HIDC, 0, stream>>>(zo, cls_out_W, cls_out_b, (float*)d_out);
}
